// K_Cache_Class_26164940767625
// MI455X (gfx1250) — compile-verified, matrix operand order checked by reading, not run on hardware
//
#include <hip/hip_runtime.h>

typedef __attribute__((ext_vector_type(16))) _Float16 v16h;
typedef __attribute__((ext_vector_type(8)))  _Float16 v8h;
typedef __attribute__((ext_vector_type(8)))  float    v8f;

#define BSZ 8
#define SEQ 8192
#define NKV 8
#define DH  128
#define ROWSTRIDE (NKV * DH)   // 1024 halves between consecutive cached positions

// One wave computes 16 scores per tile via v_wmma_f32_16x16x32_f16:
//   A (16x32 f16)  = 16 K-cache rows x 32-dim chunk of D
//   B (32x16 f16)  = q chunk broadcast into all 16 columns
//   D column n (any n) = the 16 scores; lanes 0 and 16 hold all of them.
__global__ __launch_bounds__(256)
void qk_bitcache_scores(const _Float16* __restrict__ Kc,
                        const _Float16* __restrict__ Q,
                        const int* __restrict__ startPos,
                        const int* __restrict__ seqLen,
                        float* __restrict__ out)
{
    const int bh   = blockIdx.x;            // b*NKV + h  (0..63)
    const int wave = threadIdx.x >> 5;      // 0..7
    const int lane = threadIdx.x & 31;
    const int mrow = lane & 15;             // row (M) id within half-group
    const int half = lane >> 4;             // 0: holds K 0..7 & 16..23 ; 1: K 8..15 & 24..31

    const int total = startPos[0] + seqLen[0];

    const int b = bh >> 3, h = bh & 7;
    const _Float16* qptr = Q  + (size_t)(b * NKV + h) * DH;
    const _Float16* kbh  = Kc + (size_t)b * SEQ * ROWSTRIDE + (size_t)h * DH;

    // Build the broadcast-q B operand for each of the 4 K-chunks (D = 4*32).
    // ISA 16-bit operand layout: lanes 0-15 take chunk-local K 0..7 (v0-3) and
    // 16..23 (v4-7); lanes 16-31 take 8..15 and 24..31.
    v16h bq[4];
#pragma unroll
    for (int kk = 0; kk < 4; ++kk) {
        v8h lo = *(const v8h*)(qptr + kk * 32 +      half * 8);
        v8h hi = *(const v8h*)(qptr + kk * 32 + 16 + half * 8);
        v16h t;
#pragma unroll
        for (int i = 0; i < 8; ++i) { t[i] = lo[i]; t[i + 8] = hi[i]; }
        bq[kk] = t;
    }

    const float scale = 0.08838834764831845f;   // 1/sqrt(128)

    // blockIdx.y covers 512 positions; each wave covers 4 tiles of 16.
    const int posWave = blockIdx.y * 512 + wave * 64;

#pragma unroll
    for (int t = 0; t < 4; ++t) {
        const int pos0 = posWave + t * 16;
        const _Float16* row = kbh + (size_t)(pos0 + mrow) * ROWSTRIDE;

        if (t < 3)  // hint next tile's row toward cache (global_prefetch_b8)
            __builtin_prefetch(row + 16 * ROWSTRIDE, 0, 3);

        // Issue ALL tile loads first (single clause, 8 b128s in flight),
        // so the WMMA chain below can run on partial loadcnt waits.
        v8h alo[4], ahi[4];
#pragma unroll
        for (int kk = 0; kk < 4; ++kk) {
            alo[kk] = *(const v8h*)(row + kk * 32 +      half * 8);
            ahi[kk] = *(const v8h*)(row + kk * 32 + 16 + half * 8);
        }

        v8f c = {};
#pragma unroll
        for (int kk = 0; kk < 4; ++kk) {
            v16h a;
#pragma unroll
            for (int i = 0; i < 8; ++i) { a[i] = alo[kk][i]; a[i + 8] = ahi[kk][i]; }
            // D = A x B + C   (accumulate over the 128-dim in 4 steps)
            c = __builtin_amdgcn_wmma_f32_16x16x32_f16(
                    /*neg_a=*/false, a, /*neg_b=*/false, bq[kk],
                    /*c_mod=*/(short)0, c, /*reuse_a=*/false, /*reuse_b=*/false);
        }

        // C/D layout: VGPR j -> M=j (lanes 0-15), M=j+8 (lanes 16-31).
        // All N columns identical (q broadcast); lanes 0 and 16 commit results.
        if (mrow == 0) {
            const int base = pos0 + half * 8;
            v8f r;
#pragma unroll
            for (int j = 0; j < 8; ++j) {
                // fp16 rounding as in the reference, then causal mask
                float s = (float)(_Float16)(c[j] * scale);
                r[j] = (base + j < total) ? s : -__builtin_inff();
            }
            *(v8f*)(out + (size_t)bh * SEQ + base) = r;
        }
    }
}

extern "C" void kernel_launch(void* const* d_in, const int* in_sizes, int n_in,
                              void* d_out, int out_size, void* d_ws, size_t ws_size,
                              hipStream_t stream)
{
    const _Float16* Kc = (const _Float16*)d_in[0];   // new_k (8,8192,8,128) fp16
    const _Float16* Q  = (const _Float16*)d_in[1];   // q     (8,1,8,128)    fp16
    const int* sp = (const int*)d_in[2];             // start_pos scalar
    const int* sl = (const int*)d_in[3];             // seqlen scalar
    float* out = (float*)d_out;                      // (8,8,1,8192)

    dim3 grid(BSZ * NKV, SEQ / 512);                 // (64, 16)
    qk_bitcache_scores<<<grid, 256, 0, stream>>>(Kc, Q, sp, sl, out);
}
